// SeFTNetwork_85968065397118
// MI455X (gfx1250) — matrix operand order repaired and checked by reading, not compiled, not run on hardware
//
#include <hip/hip_runtime.h>
#include <hip/hip_bf16.h>

// ---------------------------------------------------------------------------
// SeFT network on gfx1250: bf16 WMMA (v_wmma_f32_16x16x32_bf16) for the
// per-token MLP chain (which is >95% of the FLOPs), VALU for softmax/reduce.
// Fixed problem shapes from setup_inputs(): B=32, L=1008, V=32, R=2048.
// ---------------------------------------------------------------------------

typedef __attribute__((ext_vector_type(16))) __bf16 v16bf;
typedef __attribute__((ext_vector_type(8)))  float  v8f;

#define NPAT   32
#define LSEQ   1008            // 63 * 16
#define NTILES ((NPAT * LSEQ) / 16)   // 2016
#define NIN    34
#define K0PAD  64              // layer-0 K padded to 2 WMMA k-blocks
#define HID    128

// ---------------------------------------------------------------------------
// Weight swizzle: W (row-major [N][Kin], f32) -> bf16 B-operand tiles.
// B-operand layout (16-bit, 32x16 K x N per tile, wave32):
//   lanes 0-15 hold K = kb*32 + {0..7, 16..23}, lanes 16-31 hold K = kb*32 + {8..15, 24..31},
//   N = nt*16 + (lane & 15); 16 bf16 per lane, stored contiguously so the
//   GEMM kernel loads each lane's operand with two b128 loads.
// ---------------------------------------------------------------------------
__global__ void prep_weights(const float* __restrict__ W, int Kin,
                             __bf16* __restrict__ dst) {
    int t    = blockIdx.x;        // kb*8 + nt
    int kb   = t >> 3;
    int nt   = t & 7;
    int lane = threadIdx.x;       // 0..31
    int n    = nt * 16 + (lane & 15);
    int kbase = kb * 32 + ((lane >> 4) & 1) * 8;
    __bf16* p = dst + ((size_t)t * 32 + lane) * 16;
#pragma unroll
    for (int e = 0; e < 16; ++e) {
        int k = kbase + e + (e >= 8 ? 8 : 0);
        float v = (k < Kin) ? W[(size_t)n * Kin + k] : 0.0f;
        p[e] = (__bf16)v;
    }
}

// ---------------------------------------------------------------------------
// Gather + featurize. One block per patient; 256 threads scan the patient's
// 64x32 block of M (2048 entries, 8 per thread), prefix-scan ranks, and emit
// 64-wide bf16 feature rows (34 real dims + zero pad).  Padded sequence
// positions get the reference's pad features: sin(0)=0, cos(0)=1, v=0, x=0.
// ---------------------------------------------------------------------------
__global__ void gather_featurize(const float* __restrict__ times,
                                 const float* __restrict__ X,
                                 const int*   __restrict__ M,
                                 __bf16* __restrict__ Feat,
                                 int* __restrict__ lens) {
    __shared__ int sc[256];
    const int b   = blockIdx.x;
    const int tid = threadIdx.x;
    const int base = b * 2048;

    int mloc[8];
    int c = 0;
#pragma unroll
    for (int i = 0; i < 8; ++i) {
        int m = M[base + tid * 8 + i];
        mloc[i] = m;
        c += (m != 0);
    }
    sc[tid] = c;
    __syncthreads();
    // inclusive scan over 256 thread-counts
    for (int off = 1; off < 256; off <<= 1) {
        int v = (tid >= off) ? sc[tid - off] : 0;
        __syncthreads();
        sc[tid] += v;
        __syncthreads();
    }
    const int total = sc[255];
    const int start = sc[tid] - c;
    if (tid == 0) lens[b] = total;

    int pos = start;
#pragma unroll
    for (int i = 0; i < 8; ++i) {
        if (mloc[i]) {
            if (pos < LSEQ) {
                int e   = tid * 8 + i;
                int row = b * 64 + (e >> 5);
                int col = e & 31;
                float t  = times[row];
                float xv = X[(size_t)row * 32 + col];
                __bf16* p = Feat + ((size_t)b * LSEQ + pos) * K0PAD;
#pragma unroll
                for (int k = 0; k < 16; ++k) {
                    float ts  = __powf(100.0f, (float)k * (1.0f / 15.0f));
                    float arg = t / ts;
                    p[k]      = (__bf16)__sinf(arg);
                    p[16 + k] = (__bf16)__cosf(arg);
                }
                p[32] = (__bf16)(float)col;
                p[33] = (__bf16)xv;
#pragma unroll
                for (int k = NIN; k < K0PAD; ++k) p[k] = (__bf16)0.0f;
            }
            ++pos;
        }
    }
    // pad rows: St=Sv=Sx=0 -> [0 x16, 1 x16, 0, 0, ...]
    for (int j = total + tid; j < LSEQ; j += 256) {
        __bf16* p = Feat + ((size_t)b * LSEQ + j) * K0PAD;
#pragma unroll
        for (int k = 0; k < 16; ++k) { p[k] = (__bf16)0.0f; p[16 + k] = (__bf16)1.0f; }
#pragma unroll
        for (int k = 32; k < K0PAD; ++k) p[k] = (__bf16)0.0f;
    }
}

// ---------------------------------------------------------------------------
// WMMA helpers
// ---------------------------------------------------------------------------
union V16 { v16bf v; uint4 q[2]; };

// A-operand (16-bit 16x32 MxK): lanes 0-15 -> M=lane, K = {0..7, 16..23};
// lanes 16-31 -> M=lane-16, K = {8..15, 24..31}.  From row-major [M][ld]
// bf16 storage this is two contiguous 16-byte loads per lane.
__device__ __forceinline__ v16bf load_a(const __bf16* row_base /* + m*ld + kb*32 + hi*8 */) {
    V16 u;
    u.q[0] = *(const uint4*)(row_base);
    u.q[1] = *(const uint4*)(row_base + 16);
    return u.v;
}

__device__ __forceinline__ v16bf load_b(const __bf16* Wb, int kb, int nt, int lane) {
    const __bf16* p = Wb + ((size_t)(kb * 8 + nt) * 32 + lane) * 16;
    V16 u;
    u.q[0] = *(const uint4*)(p);
    u.q[1] = *(const uint4*)(p + 8);
    return u.v;
}

// ---------------------------------------------------------------------------
// MLP chain: Feat(16x64) -> relu(W0) -> relu(W1) -> W2 = enc -> Wk = k.
// 4 waves per block, one independent 16-token tile per wave (tiles never
// cross patients since L = 63*16).  C/D tiles are re-laid to A-operand form
// via a per-wave 16x128 bf16 LDS buffer.
// ---------------------------------------------------------------------------
__global__ void __launch_bounds__(128)
mlp_wmma(const __bf16* __restrict__ Feat,
         const float* __restrict__ b0, const float* __restrict__ b1,
         const float* __restrict__ b2, const float* __restrict__ bk,
         const __bf16* __restrict__ W0b, const __bf16* __restrict__ W1b,
         const __bf16* __restrict__ W2b, const __bf16* __restrict__ Wkb,
         __bf16* __restrict__ Enc, __bf16* __restrict__ Kbuf) {
    __shared__ __bf16 smA[4][16 * HID];
    __shared__ __bf16 smB[4][16 * HID];

    const int wave   = threadIdx.x >> 5;
    const int lane   = threadIdx.x & 31;
    const int tile   = blockIdx.x * 4 + wave;   // < NTILES
    const int laneHi = (lane >> 4) & 1;
    const int l15    = lane & 15;

    __bf16* hA = smA[wave];
    __bf16* hB = smB[wave];

    // ---- layer 0: Feat (K=64) -> hA ----
    const __bf16* fb = Feat + (size_t)tile * 16 * K0PAD + (size_t)l15 * K0PAD + laneHi * 8;
    __builtin_prefetch(Feat + (size_t)(tile + 4) * 16 * K0PAD, 0, 0);
    v16bf a0[2];
#pragma unroll
    for (int kb = 0; kb < 2; ++kb) a0[kb] = load_a(fb + kb * 32);
#pragma unroll
    for (int nt = 0; nt < 8; ++nt) {
        v8f c = {};
#pragma unroll
        for (int kb = 0; kb < 2; ++kb) {
            v16bf bt = load_b(W0b, kb, nt, lane);
            c = __builtin_amdgcn_wmma_f32_16x16x32_bf16(false, a0[kb], false, bt,
                                                        (short)0, c, false, false);
        }
        float bias = b0[nt * 16 + l15];
#pragma unroll
        for (int r = 0; r < 8; ++r) {
            float v = c[r] + bias;
            v = v > 0.0f ? v : 0.0f;
            hA[(r + laneHi * 8) * HID + nt * 16 + l15] = (__bf16)v;
        }
    }
    __syncthreads();

    v16bf a[4];
    // ---- layer 1: hA -> hB (relu) ----
#pragma unroll
    for (int kb = 0; kb < 4; ++kb) a[kb] = load_a(hA + l15 * HID + kb * 32 + laneHi * 8);
#pragma unroll
    for (int nt = 0; nt < 8; ++nt) {
        v8f c = {};
#pragma unroll
        for (int kb = 0; kb < 4; ++kb) {
            v16bf bt = load_b(W1b, kb, nt, lane);
            c = __builtin_amdgcn_wmma_f32_16x16x32_bf16(false, a[kb], false, bt,
                                                        (short)0, c, false, false);
        }
        float bias = b1[nt * 16 + l15];
#pragma unroll
        for (int r = 0; r < 8; ++r) {
            float v = c[r] + bias;
            v = v > 0.0f ? v : 0.0f;
            hB[(r + laneHi * 8) * HID + nt * 16 + l15] = (__bf16)v;
        }
    }
    __syncthreads();

    // ---- layer 2: hB -> enc (hA + global Enc, no relu) ----
#pragma unroll
    for (int kb = 0; kb < 4; ++kb) a[kb] = load_a(hB + l15 * HID + kb * 32 + laneHi * 8);
#pragma unroll
    for (int nt = 0; nt < 8; ++nt) {
        v8f c = {};
#pragma unroll
        for (int kb = 0; kb < 4; ++kb) {
            v16bf bt = load_b(W2b, kb, nt, lane);
            c = __builtin_amdgcn_wmma_f32_16x16x32_bf16(false, a[kb], false, bt,
                                                        (short)0, c, false, false);
        }
        float bias = b2[nt * 16 + l15];
#pragma unroll
        for (int r = 0; r < 8; ++r) {
            float v = c[r] + bias;
            int row = r + laneHi * 8;
            __bf16 bv = (__bf16)v;
            hA[row * HID + nt * 16 + l15] = bv;
            Enc[((size_t)tile * 16 + row) * HID + nt * 16 + l15] = bv;
        }
    }
    __syncthreads();

    // ---- k projection: enc(hA) -> Kbuf (no relu) ----
#pragma unroll
    for (int kb = 0; kb < 4; ++kb) a[kb] = load_a(hA + l15 * HID + kb * 32 + laneHi * 8);
#pragma unroll
    for (int nt = 0; nt < 8; ++nt) {
        v8f c = {};
#pragma unroll
        for (int kb = 0; kb < 4; ++kb) {
            v16bf bt = load_b(Wkb, kb, nt, lane);
            c = __builtin_amdgcn_wmma_f32_16x16x32_bf16(false, a[kb], false, bt,
                                                        (short)0, c, false, false);
        }
        float bias = bk[nt * 16 + l15];
#pragma unroll
        for (int r = 0; r < 8; ++r) {
            float v = c[r] + bias;
            int row = r + laneHi * 8;
            Kbuf[((size_t)tile * 16 + row) * HID + nt * 16 + l15] = (__bf16)v;
        }
    }
}

// ---------------------------------------------------------------------------
// q of the last sequence position only (reference uses attn[:, :, -1, :]).
// Tiny: 32 x (128 outputs x 128 MACs) -> plain VALU kernel.
// ---------------------------------------------------------------------------
__global__ void qlast_kernel(const __bf16* __restrict__ Enc,
                             const float* __restrict__ Wq,
                             const float* __restrict__ bq,
                             float* __restrict__ qlast) {
    int b = blockIdx.x;
    int n = threadIdx.x;   // 0..127
    const __bf16* e = Enc + ((size_t)b * LSEQ + (LSEQ - 1)) * HID;
    float acc = bq[n];
#pragma unroll 8
    for (int k = 0; k < HID; ++k) acc += (float)e[k] * Wq[(size_t)n * HID + k];
    qlast[b * HID + n] = acc;
}

// ---------------------------------------------------------------------------
// Attention: per (b, h) block.  scores -> masked softmax over 1008 keys ->
// weighted sum of enc (full 128 dims) -> out[b, h*128 + e].
// ---------------------------------------------------------------------------
__global__ void __launch_bounds__(256)
attn_kernel(const __bf16* __restrict__ Enc, const __bf16* __restrict__ Kbuf,
            const float* __restrict__ qlast, const int* __restrict__ lens,
            float* __restrict__ out) {
    __shared__ float s[LSEQ];
    __shared__ float qh[32];
    __shared__ float red[256];
    __shared__ float acc2[HID];

    const int bh = blockIdx.x;
    const int b  = bh >> 2;
    const int h  = bh & 3;
    const int tid = threadIdx.x;
    const int len = lens[b];

    if (tid < 32) qh[tid] = qlast[b * HID + h * 32 + tid];
    __syncthreads();

    float lmax = -1e30f;
    for (int m = tid; m < LSEQ; m += 256) {
        const __bf16* kp = Kbuf + ((size_t)b * LSEQ + m) * HID + h * 32;
        float d = 0.0f;
#pragma unroll
        for (int i = 0; i < 32; ++i) d += qh[i] * (float)kp[i];
        d *= 0.17677669529663687f;          // 1/sqrt(32)
        if (m >= len) d = -1e30f;
        s[m] = d;
        lmax = fmaxf(lmax, d);
    }
    red[tid] = lmax;
    __syncthreads();
    for (int off = 128; off > 0; off >>= 1) {
        if (tid < off) red[tid] = fmaxf(red[tid], red[tid + off]);
        __syncthreads();
    }
    const float mx = red[0];
    __syncthreads();

    float lsum = 0.0f;
    for (int m = tid; m < LSEQ; m += 256) {
        float p = __expf(s[m] - mx);
        s[m] = p;
        lsum += p;
    }
    red[tid] = lsum;
    __syncthreads();
    for (int off = 128; off > 0; off >>= 1) {
        if (tid < off) red[tid] += red[tid + off];
        __syncthreads();
    }
    const float inv = 1.0f / red[0];
    __syncthreads();

    const int e    = tid & 127;
    const int half = tid >> 7;
    float a = 0.0f;
    for (int m = half * (LSEQ / 2); m < (half + 1) * (LSEQ / 2); ++m)
        a += s[m] * (float)Enc[((size_t)b * LSEQ + m) * HID + e];
    if (half == 0) acc2[e] = a;
    __syncthreads();
    if (half == 1) out[(size_t)b * 512 + h * HID + e] = (acc2[e] + a) * inv;
}

// ---------------------------------------------------------------------------
// Launch.  Workspace layout (bytes, all offsets 256-aligned):
//   Feat  bf16 [32256][64]   @ 0         (4,128,768)
//   Enc   bf16 [32256][128]  @ 4,128,768 (8,257,536)
//   Kbuf  bf16 [32256][128]  @12,386,304 (8,257,536)
//   W0b/W1b/W2b/Wkb bf16     @20,643,840 (16K + 3*32K)
//   qlast f32  [32][128]     @20,758,528+32768 ...
//   lens  int  [32]
// Total ~20.9 MB.
// ---------------------------------------------------------------------------
extern "C" void kernel_launch(void* const* d_in, const int* in_sizes, int n_in,
                              void* d_out, int out_size, void* d_ws, size_t ws_size,
                              hipStream_t stream) {
    const float* times = (const float*)d_in[0];
    const float* X     = (const float*)d_in[2];
    const int*   M     = (const int*)  d_in[3];
    const float* W0 = (const float*)d_in[6];
    const float* b0 = (const float*)d_in[7];
    const float* W1 = (const float*)d_in[8];
    const float* b1 = (const float*)d_in[9];
    const float* W2 = (const float*)d_in[10];
    const float* b2 = (const float*)d_in[11];
    const float* Wq = (const float*)d_in[12];
    const float* bq = (const float*)d_in[13];
    const float* Wk = (const float*)d_in[14];
    const float* bk = (const float*)d_in[15];
    float* out = (float*)d_out;

    char* ws = (char*)d_ws;
    size_t off = 0;
    __bf16* Feat = (__bf16*)(ws + off); off += (size_t)NPAT * LSEQ * K0PAD * 2;
    __bf16* Enc  = (__bf16*)(ws + off); off += (size_t)NPAT * LSEQ * HID * 2;
    __bf16* Kbuf = (__bf16*)(ws + off); off += (size_t)NPAT * LSEQ * HID * 2;
    __bf16* W0b  = (__bf16*)(ws + off); off += (size_t)2 * 8 * 32 * 16 * 2;
    __bf16* W1b  = (__bf16*)(ws + off); off += (size_t)4 * 8 * 32 * 16 * 2;
    __bf16* W2b  = (__bf16*)(ws + off); off += (size_t)4 * 8 * 32 * 16 * 2;
    __bf16* Wkb  = (__bf16*)(ws + off); off += (size_t)4 * 8 * 32 * 16 * 2;
    float*  qlast = (float*)(ws + off); off += (size_t)NPAT * HID * 4;
    int*    lens  = (int*)(ws + off);   off += 256;

    prep_weights<<<16, 32, 0, stream>>>(W0, NIN, W0b);
    prep_weights<<<32, 32, 0, stream>>>(W1, HID, W1b);
    prep_weights<<<32, 32, 0, stream>>>(W2, HID, W2b);
    prep_weights<<<32, 32, 0, stream>>>(Wk, HID, Wkb);

    gather_featurize<<<NPAT, 256, 0, stream>>>(times, X, M, Feat, lens);

    mlp_wmma<<<NTILES / 4, 128, 0, stream>>>(Feat, b0, b1, b2, bk,
                                             W0b, W1b, W2b, Wkb, Enc, Kbuf);

    qlast_kernel<<<NPAT, 128, 0, stream>>>(Enc, Wq, bq, qlast);

    attn_kernel<<<NPAT * 4, 256, 0, stream>>>(Enc, Kbuf, qlast, lens, out);
}
